// SemanticModule_23536420782602
// MI455X (gfx1250) — matrix operand order repaired and checked by reading, not compiled, not run on hardware
//
#include <hip/hip_runtime.h>
#include <hip/hip_bf16.h>
#include <stddef.h>

// ---------------------------------------------------------------------------
// Problem constants (from reference): B=8, S=4096, DIM=128, VOCAB=8192
// tokens = B*S = 32768, K = 2*DIM = 256
// d_out layout: proposal (complex64 -> 2 floats each) [8*4096*128*2] floats,
//               salience [32768] floats, vq_loss [1] float.
// ---------------------------------------------------------------------------
#define NTOK   32768
#define KDIM   256
#define HALF   128
#define VOCAB  8192
#define SAL_OFF  8388608u   // 8*4096*128*2
#define LOSS_OFF 8421376u   // SAL_OFF + 32768

typedef __attribute__((ext_vector_type(16))) __bf16          v16bf;
typedef __attribute__((ext_vector_type(8)))  float           v8f;
typedef __attribute__((ext_vector_type(8)))  unsigned short  v8us;
typedef __attribute__((ext_vector_type(4)))  float           v4f;

// Bit-exact fp32 -> bf16 round-to-nearest-even (finite inputs).
__device__ __forceinline__ unsigned short f32_to_bf16(float x) {
    unsigned u = __float_as_uint(x);
    u += 0x7fffu + ((u >> 16) & 1u);
    return (unsigned short)(u >> 16);
}
__device__ __forceinline__ float bf16_to_f32(unsigned short h) {
    return __uint_as_float(((unsigned)h) << 16);
}

// Union to assemble WMMA operand registers from two 16-byte loads.
union V16 {
    v16bf v;
    v8us  h[2];
    unsigned short u[16];
};

// Convert 8 contiguous fp32 values at p into hi/lo bf16 split (dst[0..7]).
__device__ __forceinline__ void cvt8(const float* p, unsigned short* hi,
                                     unsigned short* lo) {
    v4f f0 = *(const v4f*)p;
    v4f f1 = *(const v4f*)(p + 4);
#pragma unroll
    for (int i = 0; i < 8; ++i) {
        float x = (i < 4) ? f0[i] : f1[i - 4];
        unsigned short h = f32_to_bf16(x);
        hi[i] = h;
        lo[i] = f32_to_bf16(x - bf16_to_f32(h));
    }
}

// ---------------------------------------------------------------------------
// Kernel 1: codebook -> bf16 hi/lo split + row squared norms.
// grid = VOCAB, block = 256 (one row per block).
// ---------------------------------------------------------------------------
__global__ __launch_bounds__(256) void prep_codebook(
    const float* __restrict__ cb, unsigned short* __restrict__ c_hi,
    unsigned short* __restrict__ c_lo, float* __restrict__ cnorm) {
    __shared__ float red[256];
    const int row = blockIdx.x;
    const int k   = threadIdx.x;
    const size_t o = (size_t)row * KDIM + k;
    float c = cb[o];
    unsigned short h = f32_to_bf16(c);
    c_hi[o] = h;
    c_lo[o] = f32_to_bf16(c - bf16_to_f32(h));
    red[k] = c * c;
    __syncthreads();
#pragma unroll
    for (int s = 128; s > 0; s >>= 1) {
        if (k < s) red[k] += red[k + s];
        __syncthreads();
    }
    if (k == 0) cnorm[row] = red[0];
}

// ---------------------------------------------------------------------------
// Kernel 2: nearest-codebook argmin via bf16 WMMA with hi/lo compensation.
//   score(t, v) = cnorm[v] - 2 * dot(z_t, c_v)
// block = 256 threads (8 wave32), 16 tokens per wave, 128 tokens per block.
// grid = NTOK / 128 = 256.
// Three INDEPENDENT fp32 accumulator tiles (hh, hl, lh) so the XDL pipe sees
// three interleavable 8-deep WMMA chains instead of one 24-deep RAW chain.
// A-matrix (16x32 bf16) ISA layout: lane<16 holds M=lane, K {0-7,16-23};
// lane>=16 holds M=lane-16, K {8-15,24-31}; 2x b128 loads per operand.
// B-matrix (32x16 bf16): lane group holds one code column, same K split as A ({0-7,16-23} / {8-15,24-31}).
// ---------------------------------------------------------------------------
__global__ __launch_bounds__(256) void argmin_kernel(
    const float* __restrict__ gw_real, const float* __restrict__ gw_imag,
    const unsigned short* __restrict__ c_hi,
    const unsigned short* __restrict__ c_lo,
    const float* __restrict__ cnorm, int* __restrict__ amin) {
    const int tid  = threadIdx.x;
    const int wave = tid >> 5;
    const int lane = tid & 31;
    const int l16  = lane & 15;
    const int hi16 = lane >> 4;  // 0: low K half, 1: high K half

    // --- Preload A (this wave's 16 tokens, full K=256, hi+lo split) -------
    const int token = blockIdx.x * 128 + wave * 16 + l16;
    const float* zr = gw_real + (size_t)token * HALF;
    const float* zi = gw_imag + (size_t)token * HALF;
    V16 a_hi[8], a_lo[8];
#pragma unroll
    for (int kc = 0; kc < 8; ++kc) {
        int k0 = kc * 32 + hi16 * 8;  // K values k0..k0+7  (VGPR 0-3)
        int k1 = k0 + 16;             // K values k1..k1+7  (VGPR 4-7)
        const float* p0 = (k0 < HALF) ? (zr + k0) : (zi + (k0 - HALF));
        const float* p1 = (k1 < HALF) ? (zr + k1) : (zi + (k1 - HALF));
        cvt8(p0, &a_hi[kc].u[0], &a_lo[kc].u[0]);
        cvt8(p1, &a_hi[kc].u[8], &a_lo[kc].u[8]);
    }

    // --- Sweep vocabulary, keep per-lane running argmin -------------------
    float best[8];
    int   bidx[8];
#pragma unroll
    for (int j = 0; j < 8; ++j) { best[j] = 3.4e38f; bidx[j] = 0; }

    for (int n0 = 0; n0 < VOCAB; n0 += 16) {
        const int code = n0 + l16;
        const unsigned short* bh_base =
            c_hi + (size_t)code * KDIM + hi16 * 8;
        const unsigned short* bl_base =
            c_lo + (size_t)code * KDIM + hi16 * 8;
        // Pull the next tile's cachelines toward the WGP while we compute.
        __builtin_prefetch(bh_base + (size_t)16 * KDIM, 0, 1);
        __builtin_prefetch(bl_base + (size_t)16 * KDIM, 0, 1);

        v8f acc_hh = {};  // ah * bh
        v8f acc_hl = {};  // ah * bl
        v8f acc_lh = {};  // al * bh
#pragma unroll
        for (int kc = 0; kc < 8; ++kc) {
            V16 bh, bl;
            bh.h[0] = *(const v8us*)(bh_base + kc * 32);
            bh.h[1] = *(const v8us*)(bh_base + kc * 32 + 16);
            bl.h[0] = *(const v8us*)(bl_base + kc * 32);
            bl.h[1] = *(const v8us*)(bl_base + kc * 32 + 16);
            // dot ~= ah*bh + ah*bl + al*bh  (al*bl ~ 2^-16 relative, dropped)
            acc_hh = __builtin_amdgcn_wmma_f32_16x16x32_bf16(
                false, a_hi[kc].v, false, bh.v, (short)0, acc_hh, false, false);
            acc_hl = __builtin_amdgcn_wmma_f32_16x16x32_bf16(
                false, a_hi[kc].v, false, bl.v, (short)0, acc_hl, false, false);
            acc_lh = __builtin_amdgcn_wmma_f32_16x16x32_bf16(
                false, a_lo[kc].v, false, bh.v, (short)0, acc_lh, false, false);
        }
        const float cn = cnorm[code];
#pragma unroll
        for (int j = 0; j < 8; ++j) {
            float dot = (acc_hh[j] + acc_hl[j]) + acc_lh[j];
            float s = cn - 2.0f * dot;
            // strict '<' keeps the lowest index on ties within this lane
            if (s < best[j]) { best[j] = s; bidx[j] = code; }
        }
    }

    // --- Cross-lane argmin within each 16-lane C-tile group ---------------
#pragma unroll
    for (int m = 1; m < 16; m <<= 1) {
#pragma unroll
        for (int j = 0; j < 8; ++j) {
            float os = __shfl_xor(best[j], m, 32);
            int   oi = __shfl_xor(bidx[j], m, 32);
            if (os < best[j] || (os == best[j] && oi < bidx[j])) {
                best[j] = os;
                bidx[j] = oi;
            }
        }
    }
    if (l16 == 0) {
        // C layout: lane l, VGPR j -> token M = j + 8*(l>>4)
        const int trow = blockIdx.x * 128 + wave * 16 + hi16 * 8;
#pragma unroll
        for (int j = 0; j < 8; ++j) amin[trow + j] = bidx[j];
    }
}

// ---------------------------------------------------------------------------
// Kernel 3: gather z_q, emit proposal (interleaved complex) + salience,
// per-token loss partial. grid = NTOK, block = 128 (one token per block).
// ---------------------------------------------------------------------------
__global__ __launch_bounds__(128) void gather_out(
    const float* __restrict__ cb, const float* __restrict__ gw_real,
    const float* __restrict__ gw_imag, const float* __restrict__ sal_w,
    const float* __restrict__ sal_b, const int* __restrict__ amin,
    float* __restrict__ out, float* __restrict__ loss_part) {
    __shared__ float sred[128];
    __shared__ float lred[128];
    const int t = blockIdx.x;
    const int d = threadIdx.x;
    const int idx = amin[t];
    const float re = cb[(size_t)idx * KDIM + d];
    const float im = cb[(size_t)idx * KDIM + HALF + d];
    // proposal: complex64 memory layout = interleaved (re, im)
    out[((size_t)t * HALF + d) * 2 + 0] = re;
    out[((size_t)t * HALF + d) * 2 + 1] = im;
    // salience partial: z_q . sal_w
    sred[d] = re * sal_w[d] + im * sal_w[HALF + d];
    // vq loss partial: ||z_q - z||^2
    const float dr = re - gw_real[(size_t)t * HALF + d];
    const float di = im - gw_imag[(size_t)t * HALF + d];
    lred[d] = dr * dr + di * di;
    __syncthreads();
#pragma unroll
    for (int s = 64; s > 0; s >>= 1) {
        if (d < s) {
            sred[d] += sred[d + s];
            lred[d] += lred[d + s];
        }
        __syncthreads();
    }
    if (d == 0) {
        out[SAL_OFF + (size_t)t] = sred[0] + sal_b[0];
        loss_part[t] = lred[0];
    }
}

// ---------------------------------------------------------------------------
// Kernel 4: deterministic fixed-order loss reduction (no fp atomics so graph
// replays are bit-identical). grid = 1, block = 256.
// ---------------------------------------------------------------------------
__global__ __launch_bounds__(256) void loss_reduce(
    const float* __restrict__ part, float* __restrict__ out) {
    __shared__ float red[256];
    const int k = threadIdx.x;
    float s = 0.0f;
    for (int i = 0; i < NTOK / 256; ++i) s += part[(size_t)k * (NTOK / 256) + i];
    red[k] = s;
    __syncthreads();
#pragma unroll
    for (int st = 128; st > 0; st >>= 1) {
        if (k < st) red[k] += red[k + st];
        __syncthreads();
    }
    if (k == 0) out[LOSS_OFF] = red[0] * (1.25f / 8388608.0f);
}

// ---------------------------------------------------------------------------
// Launch: inputs are {gw_real, gw_imag, codebook, sal_w, sal_b} (all fp32).
// Workspace layout (bytes):
//   [0,        4194304)  c_hi   (VOCAB*256 bf16)
//   [4194304,  8388608)  c_lo   (VOCAB*256 bf16)
//   [8388608,  8421376)  cnorm  (VOCAB f32)
//   [8421376,  8552448)  amin   (NTOK i32)
//   [8552448,  8683520)  loss_part (NTOK f32)
// ---------------------------------------------------------------------------
extern "C" void kernel_launch(void* const* d_in, const int* in_sizes, int n_in,
                              void* d_out, int out_size, void* d_ws,
                              size_t ws_size, hipStream_t stream) {
    const float* gw_real  = (const float*)d_in[0];
    const float* gw_imag  = (const float*)d_in[1];
    const float* codebook = (const float*)d_in[2];
    const float* sal_w    = (const float*)d_in[3];
    const float* sal_b    = (const float*)d_in[4];
    float* out = (float*)d_out;

    char* ws = (char*)d_ws;
    unsigned short* c_hi  = (unsigned short*)(ws);
    unsigned short* c_lo  = (unsigned short*)(ws + 4194304);
    float*          cnorm = (float*)(ws + 8388608);
    int*            amin  = (int*)(ws + 8421376);
    float*          lpart = (float*)(ws + 8552448);

    prep_codebook<<<VOCAB, 256, 0, stream>>>(codebook, c_hi, c_lo, cnorm);
    argmin_kernel<<<NTOK / 128, 256, 0, stream>>>(gw_real, gw_imag, c_hi, c_lo,
                                                  cnorm, amin);
    gather_out<<<NTOK, 128, 0, stream>>>(codebook, gw_real, gw_imag, sal_w,
                                         sal_b, amin, out, lpart);
    loss_reduce<<<1, 256, 0, stream>>>(lpart, out);
}
